// LocalAttention_75642964017710
// MI455X (gfx1250) — matrix operand order repaired and checked by reading, not compiled, not run on hardware
//
#include <hip/hip_runtime.h>

// ---------------- types / helpers ----------------
typedef __attribute__((ext_vector_type(4)))  unsigned int   u32x4;
typedef __attribute__((ext_vector_type(4)))  int            i32x4;
typedef __attribute__((ext_vector_type(8)))  int            i32x8;
typedef __attribute__((ext_vector_type(4)))  unsigned short us4;
typedef __attribute__((ext_vector_type(8)))  float          v8f;
typedef __attribute__((ext_vector_type(16))) __bf16         v16bf;

struct u32x8s { u32x4 a, b; };

static __device__ __forceinline__ v16bf make_frag(u32x4 lo, u32x4 hi) {
    u32x8s t{lo, hi};
    return __builtin_bit_cast(v16bf, t);
}

// f32 -> bf16 (round to nearest even), raw u16
static __device__ __forceinline__ unsigned short f2bf(float x) {
    unsigned int u = __builtin_bit_cast(unsigned int, x);
    unsigned int r = (u + 0x7FFFu + ((u >> 16) & 1u)) >> 16;
    return (unsigned short)r;
}

static constexpr int S_LEN  = 4096;
static constexpr int DMODEL = 1024;
static constexpr int NHEADS = 16;
static constexpr int DHEAD  = 64;

#if defined(__HIP_DEVICE_COMPILE__) && __has_builtin(__builtin_amdgcn_tensor_load_to_lds) && \
    __has_builtin(__builtin_amdgcn_s_wait_tensorcnt)
#define USE_TDM 1
#else
#define USE_TDM 0
#endif

// ================= bulk f32 -> bf16 conversion =================
__global__ __launch_bounds__(256) void cvt_f32_bf16(const float* __restrict__ in,
                                                    unsigned short* __restrict__ out, int n) {
    int i = (blockIdx.x * 256 + threadIdx.x) * 8;
    if (i < n) {
        float4 a = *(const float4*)(in + i);
        float4 b = *(const float4*)(in + i + 4);
        *(us4*)(out + i)     = (us4){f2bf(a.x), f2bf(a.y), f2bf(a.z), f2bf(a.w)};
        *(us4*)(out + i + 4) = (us4){f2bf(b.x), f2bf(b.y), f2bf(b.z), f2bf(b.w)};
    }
}

// ================= GEMM: Out[S,1024] = A[S,1024]bf16 @ W[1024,1024]bf16^T + bias ===========
// Block tile 128(M) x 64(N), K-chunk 32, 8 waves as 4(M) x 2(N); wave -> 32x32 (2x2 WMMA tiles).
static constexpr int MBLK   = 128;
static constexpr int NBLK   = 64;
static constexpr int KC     = 32;
static constexpr int NKC    = DMODEL / KC;
static constexpr int APITCH = 40;   // bf16/row in LDS (32 data + 8 pad) -> 16B-aligned rows

#if USE_TDM
// Issue one TDM 2D tile load: tile_rows x 32 bf16 from a [tensor_rows x 1024] bf16 tensor,
// LDS destination padded to APITCH via D# pad fields (interval 16 DWORDs, amount 4 DWORDs).
static __device__ __forceinline__ void tdm_issue(const unsigned short* gptr,
                                                 const unsigned short* lptr,
                                                 unsigned tensor_rows, unsigned tile_rows) {
    unsigned long long ga = (unsigned long long)(uintptr_t)gptr;
    u32x4 g0 = {0u, 0u, 0u, 0u};
    g0[0] = 1u;                                   // count=1 (valid user descriptor)
    g0[1] = (unsigned)(uintptr_t)lptr;            // lds_addr (byte offset)
    g0[2] = (unsigned)ga;                         // global_addr[31:0]
    g0[3] = (unsigned)((ga >> 32) & 0x01FFFFFFu) | 0x80000000u;   // global_addr[56:32] | type=2
    i32x8 g1 = {0, 0, 0, 0, 0, 0, 0, 0};
    // data_size=1(2B) | pad_enable | pad_interval=3(16 DW=64B) | pad_amount=3(4 DW=16B)
    g1[0] = (int)((1u << 16) | (1u << 20) | (3u << 22) | (3u << 25));
    g1[1] = (int)((DMODEL & 0xFFFFu) << 16);                      // tensor_dim0[15:0]
    g1[2] = (int)((DMODEL >> 16) | ((tensor_rows & 0xFFFFu) << 16)); // dim0[31:16] | dim1[15:0]
    g1[3] = (int)(((tensor_rows >> 16) & 0xFFFFu) | (32u << 16)); // dim1[31:16] | tile_dim0=32
    g1[4] = (int)(tile_rows & 0xFFFFu);                           // tile_dim1 (tile_dim2=0)
    g1[5] = (int)DMODEL;                                          // tensor_dim0_stride[31:0]
    i32x4 z4 = {0, 0, 0, 0};
    i32x8 z8 = {0, 0, 0, 0, 0, 0, 0, 0};
    __builtin_amdgcn_tensor_load_to_lds(g0, g1, z4, z4, z8, 0);   // 6-arg toolchain form
}
#endif

template <bool OUT_F32>
__global__ __launch_bounds__(256) void gemm_xwT(const unsigned short* __restrict__ A,
                                                const unsigned short* __restrict__ W,
                                                const float* __restrict__ bias,
                                                void* __restrict__ Outp) {
    __shared__ unsigned short Asm[2][MBLK * APITCH];   // 2 x 10240 B
    __shared__ unsigned short Wsm[2][NBLK * APITCH];   // 2 x  5120 B

    const int tid  = threadIdx.x;
    const int lane = tid & 31;
    const int w    = tid >> 5;
    const int hi   = (lane >> 4) & 1;
    const int l16  = lane & 15;
    const int wm   = w >> 1;          // 0..3 -> 32-row strip
    const int wn   = w & 1;           // 0..1 -> 32-col strip
    const int mbase = blockIdx.y * MBLK;
    const int nbase = blockIdx.x * NBLK;

    v8f acc[2][2];
    #pragma unroll
    for (int mt = 0; mt < 2; ++mt)
        #pragma unroll
        for (int nt = 0; nt < 2; ++nt)
            #pragma unroll
            for (int j = 0; j < 8; ++j) acc[mt][nt][j] = 0.0f;

#if USE_TDM
    if (w == 0) {   // prologue: chunk 0 -> buffer 0
        tdm_issue(A + (size_t)mbase * DMODEL, &Asm[0][0], S_LEN, MBLK);
        tdm_issue(W + (size_t)nbase * DMODEL, &Wsm[0][0], DMODEL, NBLK);
    }
#endif

    for (int kc = 0; kc < NKC; ++kc) {
        const int cur = kc & 1;
#if USE_TDM
        if (w == 0) {
            if (kc + 1 < NKC) {   // prefetch next chunk into other buffer
                tdm_issue(A + (size_t)mbase * DMODEL + (kc + 1) * KC, &Asm[cur ^ 1][0], S_LEN, MBLK);
                tdm_issue(W + (size_t)nbase * DMODEL + (kc + 1) * KC, &Wsm[cur ^ 1][0], DMODEL, NBLK);
                __builtin_amdgcn_s_wait_tensorcnt((short)2);   // chunk kc's pair complete
            } else {
                __builtin_amdgcn_s_wait_tensorcnt((short)0);
            }
        }
        __syncthreads();
#else
        __syncthreads();
        #pragma unroll
        for (int i = 0; i < 2; ++i) {           // A: 128x32 = 512 b128 groups
            int g = tid * 2 + i;
            int row = g >> 2, kcol = (g & 3) * 8;
            u32x4 d = *(const u32x4*)(A + (size_t)(mbase + row) * DMODEL + kc * KC + kcol);
            *(u32x4*)&Asm[cur][row * APITCH + kcol] = d;
        }
        {                                        // W: 64x32 = 256 b128 groups
            int row = tid >> 2, kcol = (tid & 3) * 8;
            u32x4 d = *(const u32x4*)(W + (size_t)(nbase + row) * DMODEL + kc * KC + kcol);
            *(u32x4*)&Wsm[cur][row * APITCH + kcol] = d;
        }
        __syncthreads();
#endif
        // ---- compute: 2 A-frags + 2 B-frags -> 4 WMMA ----
        v16bf af[2], bf[2];
        #pragma unroll
        for (int mt = 0; mt < 2; ++mt) {
            const int arow = wm * 32 + mt * 16 + l16;
            const int akb  = hi ? 8 : 0;
            af[mt] = make_frag(*(const u32x4*)&Asm[cur][arow * APITCH + akb],
                               *(const u32x4*)&Asm[cur][arow * APITCH + akb + 16]);
        }
        #pragma unroll
        for (int nt = 0; nt < 2; ++nt) {
            const int brow = wn * 32 + nt * 16 + l16;
            const int bkb  = hi ? 8 : 0;
            bf[nt] = make_frag(*(const u32x4*)&Wsm[cur][brow * APITCH + bkb],
                               *(const u32x4*)&Wsm[cur][brow * APITCH + bkb + 16]);
        }
        #pragma unroll
        for (int mt = 0; mt < 2; ++mt)
            #pragma unroll
            for (int nt = 0; nt < 2; ++nt)
                acc[mt][nt] = __builtin_amdgcn_wmma_f32_16x16x32_bf16(
                    false, af[mt], false, bf[nt], (short)0, acc[mt][nt], false, false);
#if USE_TDM
        __syncthreads();   // all reads of buf[cur] done before TDM overwrites it next iter
#endif
    }

    // ---- epilogue: C layout M = r + 8*hi, N = l16 ----
    #pragma unroll
    for (int mt = 0; mt < 2; ++mt)
        #pragma unroll
        for (int nt = 0; nt < 2; ++nt)
            #pragma unroll
            for (int r = 0; r < 8; ++r) {
                int row = mbase + wm * 32 + mt * 16 + r + hi * 8;
                int col = nbase + wn * 32 + nt * 16 + l16;
                float val = acc[mt][nt][r] + bias[col];
                if constexpr (OUT_F32)
                    ((float*)Outp)[(size_t)row * DMODEL + col] = val;
                else
                    ((unsigned short*)Outp)[(size_t)row * DMODEL + col] = f2bf(val);
            }
}

// ================= Local windowed attention =================
// Block: 1 head x 8 query tiles (128 rows). Wave w: query tile qt = qg*8 + w, streams 5
// key-chunks of 32 (tiles qt-4..qt+5). Softmax needs no running max: scaled scores are
// O(+-10) so exp2 cannot overflow f32; masked lanes contribute exactly 0.
static constexpr int KSPAN = 272;   // 17 staged key tiles per block
static constexpr int KPAD  = 280;   // mult of 8 -> 16B-aligned Vt rows
static constexpr float SCALE_LOG2E = 0.125f * 1.4426950408889634f; // (1/sqrt(64))*log2(e)

__global__ __launch_bounds__(256) void local_attn(const unsigned short* __restrict__ q,
                                                  const unsigned short* __restrict__ k,
                                                  const unsigned short* __restrict__ v,
                                                  unsigned short* __restrict__ ctx) {
    __shared__ unsigned short Vt[DHEAD * KPAD];     // V transposed [feature][key], 35840 B
    __shared__ unsigned short attnbuf[8][16 * 32];  // per-wave P tile staging, 8192 B

    const int tid  = threadIdx.x;
    const int lane = tid & 31;
    const int w    = tid >> 5;
    const int hi   = (lane >> 4) & 1;
    const int l16  = lane & 15;
    const int h    = blockIdx.y;
    const int qg   = blockIdx.x;
    const int qt   = qg * 8 + w;
    const int kt0  = qg * 8 - 4;

    // ---- stage V window transposed into LDS ----
    for (int it = 0; it < 9; ++it) {
        int id = tid + it * 256;                    // 272 rows x 8 feature-octets
        if (id < KSPAN * 8) {
            int row = id >> 3;
            int fo  = (id & 7) * 8;
            int key = kt0 * 16 + row;
            u32x4 d = {0u, 0u, 0u, 0u};
            if (key >= 0 && key < S_LEN)
                d = *(const u32x4*)(v + (size_t)key * DMODEL + h * DHEAD + fo);
            const unsigned short* ds = (const unsigned short*)&d;
            #pragma unroll
            for (int j = 0; j < 8; ++j) Vt[(fo + j) * KPAD + row] = ds[j];
        }
    }
    __syncthreads();

    // ---- Q A-fragments (16x64) from global ----
    v16bf aq[2];
    const int qrow_l = qt * 16 + l16;
    #pragma unroll
    for (int c = 0; c < 2; ++c) {
        const unsigned short* p = q + (size_t)qrow_l * DMODEL + h * DHEAD + c * 32 + (hi ? 8 : 0);
        aq[c] = make_frag(*(const u32x4*)p, *(const u32x4*)(p + 16));
    }

    float rowsum[8];
    v8f cacc[4];
    #pragma unroll
    for (int r = 0; r < 8; ++r) rowsum[r] = 0.0f;
    #pragma unroll
    for (int nt = 0; nt < 4; ++nt)
        #pragma unroll
        for (int j = 0; j < 8; ++j) cacc[nt][j] = 0.0f;

    for (int cc = 0; cc < 5; ++cc) {
        const int j0 = (qt - 4) * 16 + cc * 32;

        // ---- scores: two 16x16 tiles of Q @ K^T ----
        v8f s[2];
        #pragma unroll
        for (int t = 0; t < 2; ++t) {
            #pragma unroll
            for (int j = 0; j < 8; ++j) s[t][j] = 0.0f;
            const int jt = j0 + t * 16;
            #pragma unroll
            for (int c = 0; c < 2; ++c) {
                int key  = jt + l16;
                int keyc = min(max(key, 0), S_LEN - 1);   // masked below
                const unsigned short* p =
                    k + (size_t)keyc * DMODEL + h * DHEAD + c * 32 + (hi ? 8 : 0);
                v16bf bk = make_frag(*(const u32x4*)p, *(const u32x4*)(p + 16));
                s[t] = __builtin_amdgcn_wmma_f32_16x16x32_bf16(
                    false, aq[c], false, bk, (short)0, s[t], false, false);
            }
        }

        // ---- mask + exp2 (no max subtraction needed), per-lane partial row sums ----
        float p2[2][8];
        #pragma unroll
        for (int t = 0; t < 2; ++t) {
            int kcol = j0 + t * 16 + l16;
            bool colok = (kcol >= 0) && (kcol < S_LEN);
            #pragma unroll
            for (int r = 0; r < 8; ++r) {
                int qrow = qt * 16 + r + hi * 8;
                int d = qrow - kcol; d = (d < 0) ? -d : d;
                float e = exp2f(s[t][r] * SCALE_LOG2E);
                p2[t][r] = (colok && d <= 64) ? e : 0.0f;
                rowsum[r] += p2[t][r];
            }
        }

        // ---- C-layout -> A-layout via per-wave LDS buffer (bf16 16x32) ----
        unsigned short* ab = attnbuf[w];
        #pragma unroll
        for (int r = 0; r < 8; ++r) {
            int m = r + hi * 8;
            ab[m * 32 + l16]      = f2bf(p2[0][r]);
            ab[m * 32 + 16 + l16] = f2bf(p2[1][r]);
        }
        const unsigned short* apr = ab + l16 * 32 + (hi ? 8 : 0);
        v16bf af = make_frag(*(const u32x4*)apr, *(const u32x4*)(apr + 16));

        // ---- ctx += P @ V (V B-frags from transposed LDS) ----
        const int koff = w * 16 + cc * 32 + (hi ? 8 : 0);
        #pragma unroll
        for (int nt = 0; nt < 4; ++nt) {
            const unsigned short* vp = &Vt[(nt * 16 + l16) * KPAD + koff];
            v16bf bfv = make_frag(*(const u32x4*)vp, *(const u32x4*)(vp + 16));
            cacc[nt] = __builtin_amdgcn_wmma_f32_16x16x32_bf16(
                false, af, false, bfv, (short)0, cacc[nt], false, false);
        }
    }

    // ---- single cross-lane reduction of row sums, then normalize + store ----
    #pragma unroll
    for (int r = 0; r < 8; ++r) {
        float sum = rowsum[r];
        #pragma unroll
        for (int x = 1; x < 16; x <<= 1) sum += __shfl_xor(sum, x, 32);
        rowsum[r] = 1.0f / sum;
    }
    #pragma unroll
    for (int nt = 0; nt < 4; ++nt)
        #pragma unroll
        for (int r = 0; r < 8; ++r) {
            int row = qt * 16 + r + hi * 8;
            int col = h * DHEAD + nt * 16 + l16;
            ctx[(size_t)row * DMODEL + col] = f2bf(cacc[nt][r] * rowsum[r]);
        }
}

// ================= host launcher =================
extern "C" void kernel_launch(void* const* d_in, const int* in_sizes, int n_in,
                              void* d_out, int out_size, void* d_ws, size_t ws_size,
                              hipStream_t stream) {
    const float* query = (const float*)d_in[0];
    const float* key_  = (const float*)d_in[1];
    const float* value = (const float*)d_in[2];
    const float* Wq = (const float*)d_in[3];
    const float* bq = (const float*)d_in[4];
    const float* Wk = (const float*)d_in[5];
    const float* bk = (const float*)d_in[6];
    const float* Wv = (const float*)d_in[7];
    const float* bv = (const float*)d_in[8];
    const float* Wo = (const float*)d_in[9];
    const float* bo = (const float*)d_in[10];

    const size_t TSZ = (size_t)S_LEN * DMODEL;    // 4M elems
    const size_t WSZ = (size_t)DMODEL * DMODEL;   // 1M elems
    unsigned short* p   = (unsigned short*)d_ws;
    unsigned short* qx  = p;            p += TSZ;  // bf16 copies of inputs/weights
    unsigned short* kx  = p;            p += TSZ;
    unsigned short* vx  = p;            p += TSZ;
    unsigned short* wqb = p;            p += WSZ;
    unsigned short* wkb = p;            p += WSZ;
    unsigned short* wvb = p;            p += WSZ;
    unsigned short* wob = p;            p += WSZ;
    unsigned short* qb  = p;            p += TSZ;  // projection outputs (bf16)
    unsigned short* kb  = p;            p += TSZ;
    unsigned short* vb  = p;            p += TSZ;
    unsigned short* cb  = p;            p += TSZ;  // attention context (bf16)

    // f32 -> bf16 pre-conversion (memory-bound, ~5us total)
    cvt_f32_bf16<<<(int)(TSZ / 8 / 256), 256, 0, stream>>>(query, qx, (int)TSZ);
    cvt_f32_bf16<<<(int)(TSZ / 8 / 256), 256, 0, stream>>>(key_,  kx, (int)TSZ);
    cvt_f32_bf16<<<(int)(TSZ / 8 / 256), 256, 0, stream>>>(value, vx, (int)TSZ);
    cvt_f32_bf16<<<(int)(WSZ / 8 / 256), 256, 0, stream>>>(Wq, wqb, (int)WSZ);
    cvt_f32_bf16<<<(int)(WSZ / 8 / 256), 256, 0, stream>>>(Wk, wkb, (int)WSZ);
    cvt_f32_bf16<<<(int)(WSZ / 8 / 256), 256, 0, stream>>>(Wv, wvb, (int)WSZ);
    cvt_f32_bf16<<<(int)(WSZ / 8 / 256), 256, 0, stream>>>(Wo, wob, (int)WSZ);

    dim3 gblk(DMODEL / NBLK, S_LEN / MBLK);       // (16, 32)
    gemm_xwT<false><<<gblk, 256, 0, stream>>>(qx, wqb, bq, qb);
    gemm_xwT<false><<<gblk, 256, 0, stream>>>(kx, wkb, bk, kb);
    gemm_xwT<false><<<gblk, 256, 0, stream>>>(vx, wvb, bv, vb);

    local_attn<<<dim3(S_LEN / (16 * 8), NHEADS), 256, 0, stream>>>(qb, kb, vb, cb);

    gemm_xwT<true><<<gblk, 256, 0, stream>>>(cb, wob, bo, (float*)d_out);
}
